// dCGF_IS_65386582114981
// MI455X (gfx1250) — matrix operand order repaired and checked by reading, not compiled, not run on hardware
//
#include <hip/hip_runtime.h>

#define T_TOK   131072
#define BCOM    2048
#define NGF     1024
#define ENCH    1024
#define ZDIM    128
#define DECH    512

typedef __attribute__((ext_vector_type(16))) __bf16 v16bf;
typedef __attribute__((ext_vector_type(8)))  float  v8f;

union BF16x16 { v16bf v; unsigned u[8]; };

static __device__ __forceinline__ v8f v8f_zero() {
  v8f z = {0.f, 0.f, 0.f, 0.f, 0.f, 0.f, 0.f, 0.f};
  return z;
}

// ---------------------------------------------------------------------------
// Async global->LDS copy (ASYNCcnt-tracked, bypasses VGPRs). 16B per lane.
static __device__ __forceinline__ void async_copy_b128(const void* gptr, void* lptr) {
  const unsigned lds_off = (unsigned)(size_t)lptr;            // low 32b = LDS offset
  const unsigned long long ga = (unsigned long long)(size_t)gptr;
  asm volatile("global_load_async_to_lds_b128 %0, %1, off"
               :: "v"(lds_off), "v"(ga)
               : "memory");
}
static __device__ __forceinline__ void wait_async0() {
  asm volatile("s_wait_asynccnt 0" ::: "memory");
}

// ---------------------------------------------------------------------------
// WMMA fragment loaders for v_wmma_f32_16x16x32_bf16 (ISA 7.12.2 layouts).
//
// A (16x32 bf16, row-major source, row stride `ld` elements):
//   lane 0-15  : M = lane,    K = {0..7, 16..23}  (2 bf16 packed per VGPR)
//   lane 16-31 : M = lane-16, K = {8..15, 24..31}
static __device__ __forceinline__ v16bf load_a_frag(const __bf16* p, int row_base,
                                                    int ld, int kbase, int lane) {
  const int m = row_base + (lane & 15);
  const int g = lane >> 4;  // K-half select
  const unsigned* q = (const unsigned*)(p + (size_t)m * ld + kbase);
  BF16x16 f;
#pragma unroll
  for (int v = 0; v < 8; ++v) {
    const int idx = (v < 4) ? (v + 4 * g) : (8 + (v - 4) + 4 * g);  // dword -> K = 2*idx
    f.u[v] = q[idx];
  }
  return f.v;
}

// B (32x16 bf16, row-major K x N source, row stride `ld`):
//   lane supplies column N = nbase+(lane&15), K = {0..7,16..23} (lanes 0-15) / {8..15,24..31} (lanes 16-31).
static __device__ __forceinline__ v16bf load_b_frag(const __bf16* p, int ld,
                                                    int kbase, int nbase, int lane) {
  const __bf16* q = p + (size_t)kbase * ld + nbase + (lane & 15);
  const int g = lane >> 4;
  BF16x16 f;
#pragma unroll
  for (int s = 0; s < 16; ++s) {
    const int k = (s < 8) ? (8 * g + s) : (16 + 8 * g + (s - 8));
    f.v[s] = q[(size_t)k * ld];
  }
  return f.v;
}

// ---------------------------------------------------------------------------
__global__ void zero_f32_kernel(float* __restrict__ p, int n) {
  int i = blockIdx.x * blockDim.x + threadIdx.x;
  if (i < n) p[i] = 0.f;
}

__global__ void convert_weights_kernel(const float* __restrict__ W1,
                                       const float* __restrict__ W2,
                                       const float* __restrict__ D1,
                                       __bf16* __restrict__ W1b,
                                       __bf16* __restrict__ W2b,
                                       __bf16* __restrict__ D1b) {
  const int i = blockIdx.x * blockDim.x + threadIdx.x;
  const int n1 = 2 * NGF * ENCH;
  const int n2 = ENCH * ZDIM;
  const int n3 = ZDIM * DECH;
  if (i < n1) W1b[i] = (__bf16)W1[i];
  if (i < n2) W2b[i] = (__bf16)W2[i];
  if (i < n3) D1b[i] = (__bf16)D1[i];
}

// ---------------------------------------------------------------------------
// w[B,1024] = segment_sum(u); sorted ids -> run-length accumulate in registers,
// flush with global_atomic_add_f32 only at run boundaries.
__global__ __launch_bounds__(256) void seg_sum_w_kernel(const float* __restrict__ u,
                                                        const int* __restrict__ seg,
                                                        float* __restrict__ w) {
  const int CHUNK = 64;
  const int t0 = blockIdx.x * CHUNK;
  const int col = threadIdx.x * 4;
  float4 acc = {0.f, 0.f, 0.f, 0.f};
  int prev = seg[t0];
  for (int t = 0; t < CHUNK; ++t) {
    const int s = seg[t0 + t];  // uniform -> scalar load
    const float4 uu = *(const float4*)(u + (size_t)(t0 + t) * NGF + col);
    if (s != prev) {
      float* dst = w + (size_t)prev * NGF + col;
      atomicAdd(dst + 0, acc.x); atomicAdd(dst + 1, acc.y);
      atomicAdd(dst + 2, acc.z); atomicAdd(dst + 3, acc.w);
      acc.x = acc.y = acc.z = acc.w = 0.f;
      prev = s;
    }
    acc.x += uu.x; acc.y += uu.y; acc.z += uu.z; acc.w += uu.w;
  }
  float* dst = w + (size_t)prev * NGF + col;
  atomicAdd(dst + 0, acc.x); atomicAdd(dst + 1, acc.y);
  atomicAdd(dst + 2, acc.z); atomicAdd(dst + 3, acc.w);
}

// ---------------------------------------------------------------------------
// Generic WMMA GEMM: C[M,N] = A_f32[M,K] @ W_bf16[K,N] + bias[N].
// W tile staged via async global->LDS copies.
__global__ __launch_bounds__(256) void gemm_bias_kernel(const float* __restrict__ A,
                                                        const __bf16* __restrict__ Wb,
                                                        const float* __restrict__ bias,
                                                        float* __restrict__ C,
                                                        int M, int N, int K) {
  __shared__ __align__(16) __bf16 sA[128 * 32];
  __shared__ __align__(16) __bf16 sB[32 * 128];
  const int tid = threadIdx.x, lane = tid & 31, wave = tid >> 5;
  const int r0 = blockIdx.x * 128, n0 = blockIdx.y * 128;

  v8f acc[8];
#pragma unroll
  for (int t = 0; t < 8; ++t) acc[t] = v8f_zero();

  for (int k0 = 0; k0 < K; k0 += 32) {
    __syncthreads();
    {  // W tile 32x128 bf16: async copy, no VGPR roundtrip
#pragma unroll
      for (int j = 0; j < 2; ++j) {
        const int idx = tid + j * 256;
        const int row = idx >> 4, c = (idx & 15) * 8;
        async_copy_b128(Wb + (size_t)(k0 + row) * N + n0 + c, sB + row * 128 + c);
      }
    }
    {  // A tile 128x32 f32 -> bf16 (needs cvt, so VGPR path)
      const int row = tid >> 1, half = tid & 1;
      const float4* src = (const float4*)(A + (size_t)(r0 + row) * K + k0 + half * 16);
      __bf16* dst = sA + row * 32 + half * 16;
#pragma unroll
      for (int i = 0; i < 4; ++i) {
        const float4 f = src[i];
        dst[i * 4 + 0] = (__bf16)f.x; dst[i * 4 + 1] = (__bf16)f.y;
        dst[i * 4 + 2] = (__bf16)f.z; dst[i * 4 + 3] = (__bf16)f.w;
      }
    }
    wait_async0();
    __syncthreads();
    v16bf bfr[8];
#pragma unroll
    for (int nt = 0; nt < 8; ++nt) bfr[nt] = load_b_frag(sB, 128, 0, nt * 16, lane);
    const v16bf a = load_a_frag(sA, wave * 16, 32, 0, lane);
#pragma unroll
    for (int nt = 0; nt < 8; ++nt)
      acc[nt] = __builtin_amdgcn_wmma_f32_16x16x32_bf16(false, a, false, bfr[nt],
                                                        (short)0, acc[nt], false, false);
  }
  const int mbase = r0 + wave * 16 + 8 * (lane >> 4);
  const int ncol = lane & 15;
#pragma unroll
  for (int nt = 0; nt < 8; ++nt) {
    const int n = n0 + nt * 16 + ncol;
    const float bb = bias[n];
#pragma unroll
    for (int v = 0; v < 8; ++v)
      C[(size_t)(mbase + v) * N + n] = acc[nt][v] + bb;
  }
}

// ---------------------------------------------------------------------------
// Fused encoder, double-buffered LDS pipeline:
//   H_chunk[128,128] = u_tile @ W1_top[:,chunk]           (WMMA bf16, K=1024)
//   H_chunk = relu(H_chunk + ww[seg])                      -> bf16 in LDS
//   Z[128,128] += H_chunk @ W2[chunk,:]                    (WMMA bf16)
//   z_tok = relu(Z + b2) -> sorted-run segment reduce -> atomics into zbuf.
// Buffer selection is done with runtime integer offsets (no LDS pointer
// arrays — constant addrspacecast initializers break ld.lld).
__global__ __launch_bounds__(256) void fused_enc_kernel(const float* __restrict__ u,
                                                        const int* __restrict__ seg,
                                                        const __bf16* __restrict__ W1b,
                                                        const __bf16* __restrict__ W2b,
                                                        const float* __restrict__ ww,
                                                        const float* __restrict__ b2,
                                                        float* __restrict__ zbuf) {
  __shared__ __align__(16) char smem[65536];
  // layout: sU(buf) = smem + buf*8192          [128][32]  bf16  (x2)
  //         sW(buf) = smem + 16384 + buf*8192  [32][128]  bf16  (x2)
  //         sH      = smem + 32768             [128][128] bf16
  //         sZ      = smem (aliases stages)    [128][128] f32
  __shared__ int sseg[128];

  const int tid = threadIdx.x, lane = tid & 31, wave = tid >> 5;
  const int r0 = blockIdx.x * 128;

  if (tid < 128) sseg[tid] = seg[r0 + tid];

  const int urow = tid >> 1, uhalf = tid & 1;
  const int uoff = urow * 32 + uhalf * 16;  // element offset within a U buffer
  const float* usrc = u + (size_t)(r0 + urow) * NGF + uhalf * 16;

  v8f zacc[8];
#pragma unroll
  for (int t = 0; t < 8; ++t) zacc[t] = v8f_zero();

  for (int nc = 0; nc < 8; ++nc) {  // ENC_H in chunks of 128
    const int n0 = nc * 128;
    v8f hacc[8];
#pragma unroll
    for (int t = 0; t < 8; ++t) hacc[t] = v8f_zero();

    {  // prologue: stage k-tile 0 into buffer 0
      __bf16* sW0 = (__bf16*)(smem + 16384);
#pragma unroll
      for (int j = 0; j < 2; ++j) {
        const int idx = tid + j * 256;
        const int row = idx >> 4, c = (idx & 15) * 8;
        async_copy_b128(W1b + (size_t)row * ENCH + n0 + c, sW0 + row * 128 + c);
      }
      __bf16* dst = (__bf16*)smem + uoff;
      const float4* src = (const float4*)usrc;
#pragma unroll
      for (int i = 0; i < 4; ++i) {
        const float4 f = src[i];
        dst[i * 4 + 0] = (__bf16)f.x; dst[i * 4 + 1] = (__bf16)f.y;
        dst[i * 4 + 2] = (__bf16)f.z; dst[i * 4 + 3] = (__bf16)f.w;
      }
      wait_async0();
    }
    __syncthreads();

    for (int kt = 0; kt < NGF / 32; ++kt) {
      const int cur = kt & 1, nxt = cur ^ 1;
      const int k1 = (kt + 1) * 32;
      const bool more = (k1 < NGF);

      float4 pf[4];
      if (more) {  // issue next tile: u -> regs, W1 -> LDS (async)
        const float4* src = (const float4*)(usrc + k1);
#pragma unroll
        for (int i = 0; i < 4; ++i) pf[i] = src[i];
        __bf16* sWn = (__bf16*)(smem + 16384 + nxt * 8192);
#pragma unroll
        for (int j = 0; j < 2; ++j) {
          const int idx = tid + j * 256;
          const int row = idx >> 4, c = (idx & 15) * 8;
          async_copy_b128(W1b + (size_t)(k1 + row) * ENCH + n0 + c,
                          sWn + row * 128 + c);
        }
      }

      {  // compute current tile: batch B frags, then 8 back-to-back WMMAs
        const __bf16* sWc = (const __bf16*)(smem + 16384 + cur * 8192);
        const __bf16* sUc = (const __bf16*)(smem + cur * 8192);
        v16bf bfr[8];
#pragma unroll
        for (int nt = 0; nt < 8; ++nt) bfr[nt] = load_b_frag(sWc, 128, 0, nt * 16, lane);
        const v16bf a = load_a_frag(sUc, wave * 16, 32, 0, lane);
#pragma unroll
        for (int nt = 0; nt < 8; ++nt)
          hacc[nt] = __builtin_amdgcn_wmma_f32_16x16x32_bf16(false, a, false, bfr[nt],
                                                             (short)0, hacc[nt], false, false);
      }

      if (more) {  // drain next u tile into LDS
        __bf16* dst = (__bf16*)(smem + nxt * 8192) + uoff;
#pragma unroll
        for (int i = 0; i < 4; ++i) {
          dst[i * 4 + 0] = (__bf16)pf[i].x; dst[i * 4 + 1] = (__bf16)pf[i].y;
          dst[i * 4 + 2] = (__bf16)pf[i].z; dst[i * 4 + 3] = (__bf16)pf[i].w;
        }
        wait_async0();
      }
      __syncthreads();
    }

    {  // epilogue: + ww[seg] (b1 folded in), relu, bf16 -> sH
      __bf16* sH = (__bf16*)(smem + 32768);
      const int mbase = wave * 16 + 8 * (lane >> 4);
      const int ncol = lane & 15;
#pragma unroll
      for (int nt = 0; nt < 8; ++nt) {
#pragma unroll
        for (int v = 0; v < 8; ++v) {
          const int m = mbase + v;
          const int n = nt * 16 + ncol;
          float h = hacc[nt][v] + ww[(size_t)sseg[m] * ENCH + n0 + n];
          h = fmaxf(h, 0.f);
          sH[m * 128 + n] = (__bf16)h;
        }
      }
    }
    __syncthreads();

    // Z accumulation: zacc += sH(A-frags) @ W2[n0+..,:]
    {
      const __bf16* sH = (const __bf16*)(smem + 32768);
#pragma unroll
      for (int kk = 0; kk < 128; kk += 32) {
        v16bf bfr[8];
#pragma unroll
        for (int zt = 0; zt < 8; ++zt) bfr[zt] = load_b_frag(W2b, ZDIM, n0 + kk, zt * 16, lane);
        const v16bf a = load_a_frag(sH, wave * 16, 128, kk, lane);
#pragma unroll
        for (int zt = 0; zt < 8; ++zt)
          zacc[zt] = __builtin_amdgcn_wmma_f32_16x16x32_bf16(false, a, false, bfr[zt],
                                                             (short)0, zacc[zt], false, false);
      }
    }
  }

  __syncthreads();
  {  // z_tok = relu(Z + b2) -> LDS f32
    float* sZ = (float*)smem;
    const int mbase = wave * 16 + 8 * (lane >> 4);
    const int ncol = lane & 15;
#pragma unroll
    for (int zt = 0; zt < 8; ++zt) {
      const float bb = b2[zt * 16 + ncol];
#pragma unroll
      for (int v = 0; v < 8; ++v)
        sZ[(mbase + v) * 128 + zt * 16 + ncol] = fmaxf(zacc[zt][v] + bb, 0.f);
    }
  }
  __syncthreads();

  // sorted-segment run-length reduce down 128 rows; one column per thread
  if (tid < 128) {
    const float* sZ = (const float*)smem;
    const int c = tid;
    float acc = 0.f;
    int prev = sseg[0];
    for (int m = 0; m < 128; ++m) {
      const int s = sseg[m];
      if (s != prev) {
        atomicAdd(&zbuf[(size_t)prev * ZDIM + c], acc);
        acc = 0.f;
        prev = s;
      }
      acc += sZ[m * 128 + c];
    }
    atomicAdd(&zbuf[(size_t)prev * ZDIM + c], acc);
  }
}

// ---------------------------------------------------------------------------
// Decoder: y = relu(z @ D1 + db1) @ D2 + db2, fused (hidden never hits HBM).
__global__ __launch_bounds__(256) void decoder_kernel(const float* __restrict__ z,
                                                      const __bf16* __restrict__ D1b,
                                                      const float* __restrict__ db1,
                                                      const float* __restrict__ D2,
                                                      const float* __restrict__ db2,
                                                      float* __restrict__ y) {
  __shared__ __align__(16) __bf16 sZ[128 * 128];
  __shared__ float sy[128];
  const int tid = threadIdx.x, lane = tid & 31, wave = tid >> 5;
  const int r0 = blockIdx.x * 128;

  for (int i = tid; i < (128 * 128) / 4; i += 256) {
    const float4 f = ((const float4*)z)[(size_t)r0 * (ZDIM / 4) + i];
    sZ[i * 4 + 0] = (__bf16)f.x; sZ[i * 4 + 1] = (__bf16)f.y;
    sZ[i * 4 + 2] = (__bf16)f.z; sZ[i * 4 + 3] = (__bf16)f.w;
  }
  if (tid < 128) sy[tid] = 0.f;
  __syncthreads();

  float yp[8] = {0.f, 0.f, 0.f, 0.f, 0.f, 0.f, 0.f, 0.f};
  const int ncol = lane & 15;

  for (int ncch = 0; ncch < 4; ++ncch) {  // DEC_H in chunks of 128
    const int n0 = ncch * 128;
    v8f acc[8];
#pragma unroll
    for (int t = 0; t < 8; ++t) acc[t] = v8f_zero();
#pragma unroll
    for (int kk = 0; kk < 128; kk += 32) {
      v16bf bfr[8];
#pragma unroll
      for (int nt = 0; nt < 8; ++nt) bfr[nt] = load_b_frag(D1b, DECH, kk, n0 + nt * 16, lane);
      const v16bf a = load_a_frag(sZ, wave * 16, 128, kk, lane);
#pragma unroll
      for (int nt = 0; nt < 8; ++nt)
        acc[nt] = __builtin_amdgcn_wmma_f32_16x16x32_bf16(false, a, false, bfr[nt],
                                                          (short)0, acc[nt], false, false);
    }
#pragma unroll
    for (int nt = 0; nt < 8; ++nt) {
      const int n = n0 + nt * 16 + ncol;
      const float d2 = D2[n];
      const float bb = db1[n];
#pragma unroll
      for (int v = 0; v < 8; ++v)
        yp[v] += fmaxf(acc[nt][v] + bb, 0.f) * d2;
    }
  }
  // 16 lanes share each output row: reduce via LDS float atomics (ds_add_f32)
  const int mbase = wave * 16 + 8 * (lane >> 4);
#pragma unroll
  for (int v = 0; v < 8; ++v) atomicAdd(&sy[mbase + v], yp[v]);
  __syncthreads();
  if (tid < 128) y[r0 + tid] = sy[tid] + db2[0];
}

// ---------------------------------------------------------------------------
extern "C" void kernel_launch(void* const* d_in, const int* in_sizes, int n_in,
                              void* d_out, int out_size, void* d_ws, size_t ws_size,
                              hipStream_t stream) {
  const float* u   = (const float*)d_in[0];
  const int*   seg = (const int*)d_in[1];
  const float* W1  = (const float*)d_in[2];
  const float* b1  = (const float*)d_in[3];
  const float* W2  = (const float*)d_in[4];
  const float* b2  = (const float*)d_in[5];
  const float* D1  = (const float*)d_in[6];
  const float* db1 = (const float*)d_in[7];
  const float* D2  = (const float*)d_in[8];
  const float* db2 = (const float*)d_in[9];
  float* y = (float*)d_out;

  char* ws = (char*)d_ws;
  const size_t OFF_W   = 0;                                      // 8 MB  f32
  const size_t OFF_WW  = OFF_W   + (size_t)BCOM * NGF * 4;       // 8 MB  f32
  const size_t OFF_Z   = OFF_WW  + (size_t)BCOM * ENCH * 4;      // 1 MB  f32
  const size_t OFF_W1B = OFF_Z   + (size_t)BCOM * ZDIM * 4;      // 4 MB  bf16
  const size_t OFF_W2B = OFF_W1B + (size_t)2 * NGF * ENCH * 2;   // 256 KB
  const size_t OFF_D1B = OFF_W2B + (size_t)ENCH * ZDIM * 2;      // 128 KB
  float*  w    = (float*)(ws + OFF_W);
  float*  wwp  = (float*)(ws + OFF_WW);
  float*  zbuf = (float*)(ws + OFF_Z);
  __bf16* W1b  = (__bf16*)(ws + OFF_W1B);
  __bf16* W2b  = (__bf16*)(ws + OFF_W2B);
  __bf16* D1b  = (__bf16*)(ws + OFF_D1B);

  {  // zero accumulation buffers (every call: graph replays must be clean)
    const int nw = BCOM * NGF;
    zero_f32_kernel<<<(nw + 255) / 256, 256, 0, stream>>>(w, nw);
    const int nz = BCOM * ZDIM;
    zero_f32_kernel<<<(nz + 255) / 256, 256, 0, stream>>>(zbuf, nz);
  }
  {  // weights -> bf16 (L2-resident thereafter)
    const int n1 = 2 * NGF * ENCH;
    convert_weights_kernel<<<(n1 + 255) / 256, 256, 0, stream>>>(W1, W2, D1, W1b, W2b, D1b);
  }
  // w = segment_sum(u)
  seg_sum_w_kernel<<<T_TOK / 64, 256, 0, stream>>>(u, seg, w);
  // ww = w @ W1_bot + b1 (folds community context + bias into one table)
  gemm_bias_kernel<<<dim3(BCOM / 128, ENCH / 128), 256, 0, stream>>>(
      w, W1b + (size_t)NGF * ENCH, b1, wwp, BCOM, ENCH, NGF);
  // fused encoder + segment pooling
  fused_enc_kernel<<<T_TOK / 128, 256, 0, stream>>>(u, seg, W1b, W2b, wwp, b2, zbuf);
  // fused decoder MLP
  decoder_kernel<<<BCOM / 128, 256, 0, stream>>>(zbuf, D1b, db1, D2, db2, y);
}
